// MultiHeadAttention_2937757630890
// MI455X (gfx1250) — matrix operand order repaired and checked by reading, not compiled, not run on hardware
//
#include <hip/hip_runtime.h>
#include <hip/hip_bf16.h>
#include <cstdint>

// ---------------------------------------------------------------------------
// MI455X (gfx1250) multi-head attention, bf16 WMMA path + TDM staging.
//   DIM=512, HEADS=8, HEAD_DIM=64, b=4, l=m=2048.
//   ~52 GFLOP / ~70MB traffic -> matrix-engine bound; all four GEMM stages on
//   v_wmma_f32_16x16x32_bf16 (f32 accum). K chunks for the flash loop are
//   staged global->LDS by the Tensor Data Mover (TENSORcnt, double buffered)
//   so DMA overlaps WMMA+softmax.
//   Scratch (d_ws): QPh [32][2048][64] bf16 | KPh same | X [8192][512] bf16.
// ---------------------------------------------------------------------------

typedef __attribute__((ext_vector_type(16))) __bf16    v16bf;
typedef __attribute__((ext_vector_type(8)))  float     v8f;
typedef __attribute__((ext_vector_type(8)))  uint32_t  v8u32;
typedef __attribute__((ext_vector_type(4)))  uint32_t  v4u;
typedef __attribute__((ext_vector_type(8)))  int       v8i;
typedef __attribute__((ext_vector_type(4)))  int       v4i;

#define L_SEQ 2048
#define M_SEQ 2048
#define HD    64
#define DIM_  512

__device__ __forceinline__ uint16_t bf16_bits(float f) {
  uint32_t u = __builtin_bit_cast(uint32_t, f);
  uint32_t r = u + 0x7FFFu + ((u >> 16) & 1u);   // round-to-nearest-even
  return (uint16_t)(r >> 16);
}

__device__ __forceinline__ float fast_exp2(float x) {
#if __has_builtin(__builtin_amdgcn_exp2f)
  return __builtin_amdgcn_exp2f(x);
#else
  return exp2f(x);
#endif
}

__device__ __forceinline__ v8f wmma_bf16(v16bf a, v16bf b, v8f c) {
  // D = A(16x32) * B(32x16) + C, f32 accumulate
  return __builtin_amdgcn_wmma_f32_16x16x32_bf16(
      /*neg_a=*/false, a, /*neg_b=*/false, b,
      /*c_mod=*/(short)0, c, /*reuse_a=*/false, /*reuse_b=*/false);
}

// A-matrix fragment (16x32 bf16, contiguous k order): row = lane&15; per half-pair v:
// K = 2v + (lane>=16 ? 16 : 0); pairs contiguous -> dword loads.
__device__ __forceinline__ v16bf load_a_frag(const uint16_t* base,
                                             int row_stride, int lane) {
  v8u32 r;
  const uint16_t* rowp = base + (size_t)(lane & 15) * row_stride;
  const int hi8 = (lane >= 16) ? 16 : 0;
#pragma unroll
  for (int v = 0; v < 8; ++v) {
    int k0 = 2 * v + hi8;
    r[v] = *reinterpret_cast<const uint32_t*>(rowp + k0);
  }
  return __builtin_bit_cast(v16bf, r);
}

// B fragment (32x16 bf16), source stored K-major per column (col-major B):
// element [col][k] at base[col*col_stride + k]; col = lane&15,
// K = 2v + p + (lane>=16 ? 16 : 0) -> paired dword loads.
__device__ __forceinline__ v16bf load_b_frag(const uint16_t* base,
                                             int col_stride, int lane) {
  v8u32 r;
  const uint16_t* colp = base + (size_t)(lane & 15) * col_stride;
  const int off = (lane >= 16) ? 16 : 0;
#pragma unroll
  for (int v = 0; v < 8; ++v)
    r[v] = *reinterpret_cast<const uint32_t*>(colp + 2 * v + off);
  return __builtin_bit_cast(v16bf, r);
}

// B fragment, source stored row-major (element [k][col] at base[k*stride+col]).
__device__ __forceinline__ v16bf load_b_frag_rowmajor(const uint16_t* base,
                                                      int row_stride, int lane) {
  const uint16_t* p0 = base + (lane & 15) +
                       ((lane >= 16) ? (size_t)16 * row_stride : 0);
  v8u32 r;
#pragma unroll
  for (int v = 0; v < 8; ++v) {
    uint32_t lo = p0[(size_t)(2 * v) * row_stride];
    uint32_t hi = p0[(size_t)(2 * v + 1) * row_stride];
    r[v] = lo | (hi << 16);
  }
  return __builtin_bit_cast(v16bf, r);
}

// ---------------------------------------------------------------------------
// TDM: 2D tile load, 2-byte elements (ISA ch.8 D# layout; not in a cluster so
// workgroup_mask = 0, zero groups 2/3 => dims 2..4 unused).
// Toolchain here exposes the 6-arg builtin (clang-23 / therock form):
//   (uint32x4 g0, int32x8 g1, int32x4, int32x4, int32x8, i32 cpol)
// ---------------------------------------------------------------------------
#if __has_builtin(__builtin_amdgcn_tensor_load_to_lds)
#define HAVE_TDM 1
#endif

__device__ __forceinline__ void tdm_load_2d_b16(uint32_t lds_addr, uint64_t gaddr,
                                                uint32_t tensor_w, uint32_t tensor_h,
                                                uint32_t tile_w, uint32_t tile_h,
                                                uint32_t row_stride_elems) {
#ifdef HAVE_TDM
  v4u g0;
  g0[0] = 1u;                                            // count=1, user mode
  g0[1] = lds_addr;                                      // LDS byte address
  g0[2] = (uint32_t)gaddr;                               // global_addr[31:0]
  g0[3] = (uint32_t)((gaddr >> 32) & 0x1FFFFFFu)         // global_addr[56:32]
          | (2u << 30);                                  // type=2 ("image")
  v8i g1;
  g1[0] = (int)(1u << 16);                               // data_size=1 (2B)
  g1[1] = (int)(tensor_w << 16);                         // tensor_dim0[15:0]
  g1[2] = (int)((tensor_w >> 16) | (tensor_h << 16));    // td0 hi | td1 lo
  g1[3] = (int)((tensor_h >> 16) | (tile_w << 16));      // td1 hi | tile_dim0
  g1[4] = (int)tile_h;                                   // tile_dim1 (dim2=0)
  g1[5] = (int)row_stride_elems;                         // tensor_dim0_stride
  g1[6] = 0;
  g1[7] = 0;
  v4i z4 = {0, 0, 0, 0};
  v8i z8 = {0, 0, 0, 0, 0, 0, 0, 0};
  __builtin_amdgcn_tensor_load_to_lds(g0, g1, z4, z4, z8, 0);
#else
  (void)lds_addr; (void)gaddr; (void)tensor_w; (void)tensor_h;
  (void)tile_w; (void)tile_h; (void)row_stride_elems;
#endif
}

// ---------------------------------------------------------------------------
// Projection: out_bf16[bn=b*8+n][row][d] = (A[8192x512] @ W[512x512])[row][d*8+n]
// Block: 256 thr (8 waves), tile 128 rows x 64 cols, K step 32.
// ---------------------------------------------------------------------------
__global__ __launch_bounds__(256) void proj_kernel(
    const float* __restrict__ A, const float* __restrict__ W,
    uint16_t* __restrict__ outp) {
  __shared__ alignas(16) uint16_t As[128 * 32];    // A tile, row-major
  __shared__ alignas(16) uint16_t BsT[64 * 32];    // W tile, col-major (K inner)

  const int lane = threadIdx.x & 31;
  const int wave = threadIdx.x >> 5;
  const int r0 = blockIdx.x * 128;
  const int c0 = blockIdx.y * 64;

  v8f acc[4];
#pragma unroll
  for (int c = 0; c < 4; ++c)
#pragma unroll
    for (int i = 0; i < 8; ++i) acc[c][i] = 0.f;

  for (int kk = 0; kk < DIM_; kk += 32) {
    {   // stage A (f32 -> bf16), 16 elems/thread
      int row = threadIdx.x >> 1, cb = (threadIdx.x & 1) * 16;
      const float* src = A + (size_t)(r0 + row) * DIM_ + kk + cb;
      uint32_t* dst = reinterpret_cast<uint32_t*>(&As[row * 32 + cb]);
#pragma unroll
      for (int i = 0; i < 16; i += 2)
        dst[i >> 1] = (uint32_t)bf16_bits(src[i]) |
                      ((uint32_t)bf16_bits(src[i + 1]) << 16);
      if (kk + 32 < DIM_)                       // global_prefetch_b8
        __builtin_prefetch(src + 32, 0, 1);
    }
    {   // stage W transposed, 8 elems/thread
      int krow = threadIdx.x >> 3, cb = (threadIdx.x & 7) * 8;
      const float* src = W + (size_t)(kk + krow) * DIM_ + c0 + cb;
#pragma unroll
      for (int i = 0; i < 8; ++i)
        BsT[(size_t)(cb + i) * 32 + krow] = bf16_bits(src[i]);
      if (kk + 32 < DIM_)
        __builtin_prefetch(src + 32 * DIM_, 0, 1);
    }
    __syncthreads();

    v16bf af = load_a_frag(&As[wave * 16 * 32], 32, lane);
#pragma unroll
    for (int c = 0; c < 4; ++c) {
      v16bf bf = load_b_frag(&BsT[c * 16 * 32], 32, lane);
      acc[c] = wmma_bf16(af, bf, acc[c]);
    }
    __syncthreads();
  }

  // scatter: column c_full = d*8 + n  ->  head-major [b*8+n][l][d]
  const int col = lane & 15;
  const int rb = (lane >= 16) ? 8 : 0;
#pragma unroll
  for (int c = 0; c < 4; ++c)
#pragma unroll
    for (int i = 0; i < 8; ++i) {
      int g = r0 + wave * 16 + i + rb;          // global row 0..8191
      int b = g >> 11, l = g & (L_SEQ - 1);
      int cf = c0 + c * 16 + col;
      int n = cf & 7, d = cf >> 3;
      outp[(((size_t)(b * 8 + n)) * L_SEQ + l) * HD + d] = bf16_bits(acc[c][i]);
    }
}

// ---------------------------------------------------------------------------
// Flash attention: per wave 16 query rows x full 64 dims, one (b,head).
// S = Q Kp^T / 8 ; online softmax (log2 domain) ; O += P Kp (ref uses kp!).
// Block: 128 thr (4 waves -> 64 query rows). K chunks (32x64 bf16, 4KB) are
// DMA'd into double-buffered LDS by the TDM; transfer of chunk i+1 overlaps
// compute on chunk i; drained with s_wait_tensorcnt.
// ---------------------------------------------------------------------------
__global__ __launch_bounds__(128) void attn_kernel(
    const uint16_t* __restrict__ qph, const uint16_t* __restrict__ kph,
    uint16_t* __restrict__ xout) {
  __shared__ alignas(16) uint16_t kp_raw[2][32 * HD]; // [buf][m'][d]
  __shared__ alignas(16) uint16_t p_buf[4][16 * 32];  // per-wave P, [row][m']

  const int lane = threadIdx.x & 31;
  const int wave = threadIdx.x >> 5;
  const int ltile = blockIdx.x & 31;
  const int bn = blockIdx.x >> 5;                     // b*8+n
  const int qrow0 = ltile * 64 + wave * 16;
  const uint16_t* qbase = qph + (size_t)bn * L_SEQ * HD;
  const uint16_t* kbase = kph + (size_t)bn * M_SEQ * HD;

  // Q fragments (persistent): K-dim 0..31 and 32..63
  v16bf qa0 = load_a_frag(qbase + (size_t)qrow0 * HD, HD, lane);
  v16bf qa1 = load_a_frag(qbase + (size_t)qrow0 * HD + 32, HD, lane);

  v8f oacc[4];
  float m_run[8], l_run[8];
#pragma unroll
  for (int i = 0; i < 8; ++i) { m_run[i] = -1e30f; l_run[i] = 0.f; }
#pragma unroll
  for (int c = 0; c < 4; ++c)
#pragma unroll
    for (int i = 0; i < 8; ++i) oacc[c][i] = 0.f;

  const float scale_log2 = 0.125f * 1.4426950408889634f;  // /sqrt(64) * log2(e)

  // prologue: TDM chunk 0 into buffer 0 (wave 0 issues; EXEC ignored by TDM)
  if (wave == 0) {
    tdm_load_2d_b16((uint32_t)(uintptr_t)&kp_raw[0][0],
                    (uint64_t)(uintptr_t)kbase,
                    HD, M_SEQ, HD, 32, HD);
    __builtin_amdgcn_s_wait_tensorcnt(0);
  }
  __syncthreads();

  for (int mc = 0; mc < M_SEQ; mc += 32) {
    const int buf = (mc >> 5) & 1;
    const uint16_t* kcur = &kp_raw[buf][0];

    // kick DMA for next chunk into the other buffer (overlaps compute)
    if (mc + 32 < M_SEQ && wave == 0)
      tdm_load_2d_b16((uint32_t)(uintptr_t)&kp_raw[buf ^ 1][0],
                      (uint64_t)(uintptr_t)(kbase + (size_t)(mc + 32) * HD),
                      HD, M_SEQ - (mc + 32), HD, 32, HD);

    // ---- S = Q K^T : two 16x16 tiles (cols mc..+15, mc+16..+31) ----
    v8f s0, s1;
#pragma unroll
    for (int i = 0; i < 8; ++i) { s0[i] = 0.f; s1[i] = 0.f; }
    {
      v16bf b00 = load_b_frag(kcur, HD, lane);               // j=0, d 0..31
      s0 = wmma_bf16(qa0, b00, s0);
      v16bf b01 = load_b_frag(kcur + 32, HD, lane);          // j=0, d 32..63
      s0 = wmma_bf16(qa1, b01, s0);
      v16bf b10 = load_b_frag(kcur + 16 * HD, HD, lane);     // j=1
      s1 = wmma_bf16(qa0, b10, s1);
      v16bf b11 = load_b_frag(kcur + 16 * HD + 32, HD, lane);
      s1 = wmma_bf16(qa1, b11, s1);
    }

    // ---- online softmax (log2 domain); row r lives in VGPR slot r%8 of
    //      the 16 lanes of one half -> xor-butterfly within 16 lanes ----
    float mx[8];
#pragma unroll
    for (int i = 0; i < 8; ++i) {
      s0[i] *= scale_log2; s1[i] *= scale_log2;
      mx[i] = fmaxf(s0[i], s1[i]);
    }
#pragma unroll
    for (int msk = 1; msk < 16; msk <<= 1)
#pragma unroll
      for (int i = 0; i < 8; ++i) mx[i] = fmaxf(mx[i], __shfl_xor(mx[i], msk, 32));

    float alpha[8];
#pragma unroll
    for (int i = 0; i < 8; ++i) {
      float mn = fmaxf(m_run[i], mx[i]);
      alpha[i] = fast_exp2(m_run[i] - mn);
      m_run[i] = mn;
    }
    float rs[8];
#pragma unroll
    for (int i = 0; i < 8; ++i) {
      float p0 = fast_exp2(s0[i] - m_run[i]);
      float p1 = fast_exp2(s1[i] - m_run[i]);
      s0[i] = p0; s1[i] = p1;
      rs[i] = p0 + p1;
    }
#pragma unroll
    for (int msk = 1; msk < 16; msk <<= 1)
#pragma unroll
      for (int i = 0; i < 8; ++i) rs[i] += __shfl_xor(rs[i], msk, 32);
#pragma unroll
    for (int i = 0; i < 8; ++i) l_run[i] = l_run[i] * alpha[i] + rs[i];
#pragma unroll
    for (int c = 0; c < 4; ++c)
#pragma unroll
      for (int i = 0; i < 8; ++i) oacc[c][i] *= alpha[i];

    // ---- P: D layout -> row-major LDS (per-wave; same-wave DS in order) ----
    {
      uint16_t* pb = p_buf[wave];
      int col = lane & 15, rb = (lane >= 16) ? 8 : 0;
#pragma unroll
      for (int i = 0; i < 8; ++i) {
        pb[(i + rb) * 32 + col]      = bf16_bits(s0[i]);
        pb[(i + rb) * 32 + col + 16] = bf16_bits(s1[i]);
      }
    }

    // ---- O += P (16x32) @ Kp (32x64): B gathered row-major from kp_raw ----
    {
      v16bf pa = load_a_frag(p_buf[wave], 32, lane);
#pragma unroll
      for (int c = 0; c < 4; ++c) {
        v16bf bb = load_b_frag_rowmajor(kcur + c * 16, HD, lane);
        oacc[c] = wmma_bf16(pa, bb, oacc[c]);
      }
    }

    // drain next-chunk DMA; barrier also fences reads of `buf` before it is
    // overwritten by the DMA issued at the top of iteration mc+32.
    if (mc + 32 < M_SEQ) {
      if (wave == 0) __builtin_amdgcn_s_wait_tensorcnt(0);
      __syncthreads();
    }
  }

  // normalize + store re-interleaved x[b][l][d*8+n] (bf16)
  const int b = bn >> 3, n = bn & 7;
  const int col = lane & 15, rb = (lane >= 16) ? 8 : 0;
  float inv[8];
#pragma unroll
  for (int i = 0; i < 8; ++i) inv[i] = 1.f / l_run[i];
#pragma unroll
  for (int c = 0; c < 4; ++c)
#pragma unroll
    for (int i = 0; i < 8; ++i) {
      int row = qrow0 + i + rb;
      int d = c * 16 + col;
      xout[((size_t)b * L_SEQ + row) * DIM_ + d * 8 + n] =
          bf16_bits(oacc[c][i] * inv[i]);
    }
}

// ---------------------------------------------------------------------------
// Final: out_f32[8192][512] = x_bf16[8192][512] @ w_o_f32[512][512]
// ---------------------------------------------------------------------------
__global__ __launch_bounds__(256) void gemm_out_kernel(
    const uint16_t* __restrict__ X, const float* __restrict__ W,
    float* __restrict__ out) {
  __shared__ alignas(16) uint16_t As[128 * 32];
  __shared__ alignas(16) uint16_t BsT[64 * 32];

  const int lane = threadIdx.x & 31;
  const int wave = threadIdx.x >> 5;
  const int r0 = blockIdx.x * 128;
  const int c0 = blockIdx.y * 64;

  v8f acc[4];
#pragma unroll
  for (int c = 0; c < 4; ++c)
#pragma unroll
    for (int i = 0; i < 8; ++i) acc[c][i] = 0.f;

  for (int kk = 0; kk < DIM_; kk += 32) {
    {   // stage X tile (already bf16): straight dword copy
      int row = threadIdx.x >> 1, cb = (threadIdx.x & 1) * 16;
      const uint32_t* src = reinterpret_cast<const uint32_t*>(
          X + (size_t)(r0 + row) * DIM_ + kk + cb);
      uint32_t* dst = reinterpret_cast<uint32_t*>(&As[row * 32 + cb]);
#pragma unroll
      for (int i = 0; i < 8; ++i) dst[i] = src[i];
      if (kk + 32 < DIM_)
        __builtin_prefetch(src + 16, 0, 1);
    }
    {   // stage w_o transposed (f32 -> bf16)
      int krow = threadIdx.x >> 3, cb = (threadIdx.x & 7) * 8;
      const float* src = W + (size_t)(kk + krow) * DIM_ + c0 + cb;
#pragma unroll
      for (int i = 0; i < 8; ++i)
        BsT[(size_t)(cb + i) * 32 + krow] = bf16_bits(src[i]);
      if (kk + 32 < DIM_)
        __builtin_prefetch(src + 32 * DIM_, 0, 1);
    }
    __syncthreads();

    v16bf af = load_a_frag(&As[wave * 16 * 32], 32, lane);
#pragma unroll
    for (int c = 0; c < 4; ++c) {
      v16bf bf = load_b_frag(&BsT[c * 16 * 32], 32, lane);
      acc[c] = wmma_bf16(af, bf, acc[c]);
    }
    __syncthreads();
  }

  const int col = lane & 15, rb = (lane >= 16) ? 8 : 0;
#pragma unroll
  for (int c = 0; c < 4; ++c)
#pragma unroll
    for (int i = 0; i < 8; ++i) {
      int g = r0 + wave * 16 + i + rb;
      int cf = c0 + c * 16 + col;
      out[(size_t)g * DIM_ + cf] = acc[c][i];
    }
}

// ---------------------------------------------------------------------------
extern "C" void kernel_launch(void* const* d_in, const int* in_sizes, int n_in,
                              void* d_out, int out_size, void* d_ws, size_t ws_size,
                              hipStream_t stream) {
  (void)in_sizes; (void)n_in; (void)out_size; (void)ws_size;
  const float* q   = (const float*)d_in[0];
  const float* k   = (const float*)d_in[1];
  // d_in[2] = v   : unused (reference computes vp but never uses it)
  const float* w_q = (const float*)d_in[3];
  const float* w_k = (const float*)d_in[4];
  // d_in[5] = w_v : unused
  const float* w_o = (const float*)d_in[6];
  float* out = (float*)d_out;

  const size_t head_elems = (size_t)4 * 8 * L_SEQ * HD;   // 4M bf16 elems
  uint16_t* qph  = (uint16_t*)d_ws;
  uint16_t* kph  = qph + head_elems;
  uint16_t* xbuf = kph + head_elems;                      // [8192][512] bf16

  dim3 gproj(8192 / 128, DIM_ / 64), b256(256);
  proj_kernel<<<gproj, b256, 0, stream>>>(q, w_q, qph);
  proj_kernel<<<gproj, b256, 0, stream>>>(k, w_k, kph);
  attn_kernel<<<dim3(32 * (L_SEQ / 64)), dim3(128), 0, stream>>>(qph, kph, xbuf);
  gemm_out_kernel<<<gproj, b256, 0, stream>>>(xbuf, w_o, out);
}
